// CrossBlock_23484881175083
// MI455X (gfx1250) — matrix operand order repaired and checked by reading, not compiled, not run on hardware
//
#include <hip/hip_runtime.h>

typedef __bf16 bf16;
typedef __attribute__((ext_vector_type(16))) __bf16 v16bf;
typedef __attribute__((ext_vector_type(8)))  __bf16 v8bf;
typedef __attribute__((ext_vector_type(8)))  float  v8f;
typedef __attribute__((ext_vector_type(4)))  float  v4f;

// ---------------------------------------------------------------------------
// Generic WMMA GEMM:  out[m, n] = ( sum_k A[m,k] * W[n,k] + bias[n] ) * scale (+ res)
// A: [M, K] (bf16 or f32, row-major, leading dim ldA)
// W: [Nw, K] bf16 row-major (torch Linear weight layout), leading dim ldW
// Block: 256 threads = 8 waves. Block tile: 128 (M) x 64 (N). Wave tile: 16 x 64.
// grid = (Nw/64, M/128, Z). Per-z strides for batched (per-head) GEMMs.
// B fragments are read directly from global W rows (two 8-element runs per lane),
// A fragments directly from global A rows (two 8-element runs per lane).
// ---------------------------------------------------------------------------
template <bool AF32>
__global__ __launch_bounds__(256) void gemm_wmma(
    const void* __restrict__ Ap, long ldA, long strideAz,
    const bf16* __restrict__ W, long ldW, long strideWz,
    const float* __restrict__ bias, float scale,
    float* __restrict__ outF, bf16* __restrict__ outB,
    const float* __restrict__ res,
    long ldO, long ostrideB, long ostrideH, int hdiv, int K)
{
  const int z    = blockIdx.z;
  const int lane = threadIdx.x & 31;
  const int w    = threadIdx.x >> 5;
  const long m0  = (long)blockIdx.y * 128 + (long)w * 16;
  const long n0  = (long)blockIdx.x * 64;

  const float* Af = (const float*)Ap + (size_t)z * strideAz;
  const bf16*  Ab = (const bf16*)Ap + (size_t)z * strideAz;
  const bf16*  Wz = W + (size_t)z * strideWz;
  const size_t ooff = (size_t)(z / hdiv) * (size_t)ostrideB +
                      (size_t)(z % hdiv) * (size_t)ostrideH;

  const long mrow = m0 + (lane & 15);
  const int  abase = (lane >> 4) * 8;     // A: lanes 16-31 shifted by K+8
  v8f acc[4] = {};

  for (int kk = 0; kk < K; kk += 32) {
    // ---- A fragment: 16x32 bf16, row mrow, runs [abase, abase+8) and [abase+16, abase+24)
    v16bf a;
    if (AF32) {
      const float* ap = Af + (size_t)mrow * ldA + kk + abase;
      v4f f0 = *(const v4f*)(ap);
      v4f f1 = *(const v4f*)(ap + 4);
      v4f f2 = *(const v4f*)(ap + 16);
      v4f f3 = *(const v4f*)(ap + 20);
#pragma unroll
      for (int i = 0; i < 4; ++i) {
        a[i]      = (bf16)f0[i];
        a[4 + i]  = (bf16)f1[i];
        a[8 + i]  = (bf16)f2[i];
        a[12 + i] = (bf16)f3[i];
      }
    } else {
      const bf16* ap = Ab + (size_t)mrow * ldA + kk + abase;
      v8bf lo = *(const v8bf*)(ap);
      v8bf hi = *(const v8bf*)(ap + 16);
#pragma unroll
      for (int i = 0; i < 8; ++i) { a[i] = lo[i]; a[8 + i] = hi[i]; }
    }

    // ---- B fragments: lane holds column n, runs [8h, 8h+8) and [8h+16, 8h+24)
    const int koff = kk + ((lane >> 4) << 3);
#pragma unroll
    for (int ct = 0; ct < 4; ++ct) {
      const bf16* bp = Wz + (size_t)(n0 + ct * 16 + (lane & 15)) * ldW + koff;
      v8bf b0 = *(const v8bf*)(bp);
      v8bf b1 = *(const v8bf*)(bp + 16);
      v16bf bfr;
#pragma unroll
      for (int i = 0; i < 8; ++i) { bfr[i] = b0[i]; bfr[8 + i] = b1[i]; }
      acc[ct] = __builtin_amdgcn_wmma_f32_16x16x32_bf16(
          false, a, false, bfr, (short)0, acc[ct], false, false);
    }
  }

  // ---- epilogue: C layout VGPR r -> row (lane>=16 ? r+8 : r), col lane&15
  const long rbase = m0 + ((lane >> 4) << 3);
#pragma unroll
  for (int ct = 0; ct < 4; ++ct) {
    const long ncol = n0 + ct * 16 + (lane & 15);
    const float bi = bias ? bias[ncol] : 0.0f;
#pragma unroll
    for (int r = 0; r < 8; ++r) {
      float v = (acc[ct][r] + bi) * scale;
      size_t o = ooff + (size_t)(rbase + r) * ldO + ncol;
      if (res)  v += res[o];
      if (outF) outF[o] = v;
      if (outB) outB[o] = (bf16)v;
    }
  }
}

// ---------------------------------------------------------------------------
// sim kernel: per (b,h): S = qk0 [2048x64] . qk1^T  -> raw scores written to
// both attn01[i,j] and attn10[j,i] (transposed via per-wave LDS tile).
// grid = (2048/64, 2048/128, 16), block 256.
// ---------------------------------------------------------------------------
__global__ __launch_bounds__(256) void sim_wmma(
    const bf16* __restrict__ qk0, const bf16* __restrict__ qk1,
    float* __restrict__ s01, float* __restrict__ s10)
{
  __shared__ float tr[8][16][17];
  const int z = blockIdx.z;
  const int b = z >> 2, h = z & 3;
  const size_t qoff = (size_t)b * 2048 * 256 + (size_t)h * 64;
  const bf16* A  = qk0 + qoff;
  const bf16* Bq = qk1 + qoff;
  const int lane = threadIdx.x & 31;
  const int w    = threadIdx.x >> 5;
  const long i0  = (long)blockIdx.y * 128 + (long)w * 16;
  const long j0  = (long)blockIdx.x * 64;
  const int abase = (lane >> 4) * 8;

  v8f acc[4] = {};
#pragma unroll
  for (int kk = 0; kk < 64; kk += 32) {
    const bf16* ap = A + (size_t)(i0 + (lane & 15)) * 256 + kk + abase;
    v8bf lo = *(const v8bf*)(ap);
    v8bf hi = *(const v8bf*)(ap + 16);
    v16bf a;
#pragma unroll
    for (int i = 0; i < 8; ++i) { a[i] = lo[i]; a[8 + i] = hi[i]; }
    const int koff = kk + ((lane >> 4) << 3);
#pragma unroll
    for (int ct = 0; ct < 4; ++ct) {
      const bf16* bp = Bq + (size_t)(j0 + ct * 16 + (lane & 15)) * 256 + koff;
      v8bf b0 = *(const v8bf*)(bp);
      v8bf b1 = *(const v8bf*)(bp + 16);
      v16bf bfr;
#pragma unroll
      for (int i = 0; i < 8; ++i) { bfr[i] = b0[i]; bfr[8 + i] = b1[i]; }
      acc[ct] = __builtin_amdgcn_wmma_f32_16x16x32_bf16(
          false, a, false, bfr, (short)0, acc[ct], false, false);
    }
  }

  const size_t obase = (size_t)z * 2048 * 2048;
  const int rb = (lane >> 4) * 8;
  for (int ct = 0; ct < 4; ++ct) {
    const int jj = ct * 16 + (lane & 15);
#pragma unroll
    for (int r = 0; r < 8; ++r) {
      s01[obase + (size_t)(i0 + rb + r) * 2048 + j0 + jj] = acc[ct][r];
      tr[w][rb + r][lane & 15] = acc[ct][r];
    }
    __syncthreads();
    const int m = lane & 15;
#pragma unroll
    for (int e = 0; e < 8; ++e) {
      const int nloc = rb + e;
      s10[obase + (size_t)(j0 + ct * 16 + nloc) * 2048 + (i0 + m)] =
          tr[w][m][nloc];
    }
    __syncthreads();
  }
}

// ---------------------------------------------------------------------------
// in-place row softmax, rows of 2048 fp32. grid = #rows, block 256.
// ---------------------------------------------------------------------------
__global__ __launch_bounds__(256) void softmax_rows(float* __restrict__ base)
{
  __shared__ float red[256];
  float* p = base + (size_t)blockIdx.x * 2048;
  const int t = threadIdx.x;
  float v[8];
#pragma unroll
  for (int k = 0; k < 8; ++k) v[k] = p[t + k * 256];
  float mx = v[0];
#pragma unroll
  for (int k = 1; k < 8; ++k) mx = fmaxf(mx, v[k]);
  red[t] = mx; __syncthreads();
  for (int s = 128; s > 0; s >>= 1) {
    if (t < s) red[t] = fmaxf(red[t], red[t + s]);
    __syncthreads();
  }
  mx = red[0]; __syncthreads();
  float sum = 0.0f;
#pragma unroll
  for (int k = 0; k < 8; ++k) { v[k] = expf(v[k] - mx); sum += v[k]; }
  red[t] = sum; __syncthreads();
  for (int s = 128; s > 0; s >>= 1) {
    if (t < s) red[t] += red[t + s];
    __syncthreads();
  }
  const float inv = 1.0f / red[0];
#pragma unroll
  for (int k = 0; k < 8; ++k) p[t + k * 256] = v[k] * inv;
}

// ---------------------------------------------------------------------------
// LayerNorm (512) + exact-erf GELU, fp32 in -> bf16 out. grid = rows, block 256.
// ---------------------------------------------------------------------------
__global__ __launch_bounds__(256) void ln_gelu(
    const float* __restrict__ hpre, const float* __restrict__ g,
    const float* __restrict__ bb, bf16* __restrict__ out)
{
  __shared__ float red[256];
  const size_t row = blockIdx.x;
  const float* p = hpre + row * 512;
  const int t = threadIdx.x;
  const float x0 = p[t], x1 = p[t + 256];
  red[t] = x0 + x1; __syncthreads();
  for (int s = 128; s > 0; s >>= 1) {
    if (t < s) red[t] += red[t + s];
    __syncthreads();
  }
  const float mean = red[0] * (1.0f / 512.0f);
  __syncthreads();
  const float d0 = x0 - mean, d1 = x1 - mean;
  red[t] = d0 * d0 + d1 * d1; __syncthreads();
  for (int s = 128; s > 0; s >>= 1) {
    if (t < s) red[t] += red[t + s];
    __syncthreads();
  }
  const float rs = rsqrtf(red[0] * (1.0f / 512.0f) + 1e-5f);
  const float y0 = d0 * rs * g[t] + bb[t];
  const float y1 = d1 * rs * g[t + 256] + bb[t + 256];
  const float g0 = 0.5f * y0 * (1.0f + erff(y0 * 0.70710678118654752f));
  const float g1 = 0.5f * y1 * (1.0f + erff(y1 * 0.70710678118654752f));
  out[row * 512 + t]       = (bf16)g0;
  out[row * 512 + t + 256] = (bf16)g1;
}

// ---------------------------------------------------------------------------
// helpers: f32->bf16 converts and per-head V transpose
// ---------------------------------------------------------------------------
__global__ void cvt_f32_bf16(const float* __restrict__ s, bf16* __restrict__ d, int n)
{
  int i = blockIdx.x * 256 + threadIdx.x;
  if (i < n) d[i] = (bf16)s[i];
}

__global__ void cvt_rows_bf16(const float* __restrict__ s, bf16* __restrict__ d)
{ // [8192,256] f32 -> bf16 rows of a [8192,512] buffer (cols 0..255)
  int i = blockIdx.x * 256 + threadIdx.x;
  int r = i >> 8, c = i & 255;
  d[(size_t)r * 512 + c] = (bf16)s[i];
}

__global__ void transpose_v(const bf16* __restrict__ s, bf16* __restrict__ d)
{ // [8192,256] (t = b*2048+n, o = h*64+dd) -> vT[(b*4+h)*131072 + dd*2048 + n]
  int i = blockIdx.x * 256 + threadIdx.x;
  int t = i >> 8, o = i & 255;
  int b = t >> 11, n = t & 2047, h = o >> 6, dd = o & 63;
  d[(size_t)(b * 4 + h) * 131072 + (size_t)dd * 2048 + n] = s[i];
}

// ---------------------------------------------------------------------------
extern "C" void kernel_launch(void* const* d_in, const int* in_sizes, int n_in,
                              void* d_out, int out_size, void* d_ws, size_t ws_size,
                              hipStream_t stream)
{
  (void)in_sizes; (void)n_in; (void)out_size; (void)ws_size;
  const float* x0  = (const float*)d_in[0];
  const float* x1  = (const float*)d_in[1];
  const float* Wqk = (const float*)d_in[2];
  const float* bqk = (const float*)d_in[3];
  const float* Wv  = (const float*)d_in[4];
  const float* bv  = (const float*)d_in[5];
  const float* Wo  = (const float*)d_in[6];
  const float* bo  = (const float*)d_in[7];
  const float* W1  = (const float*)d_in[8];
  const float* b1  = (const float*)d_in[9];
  const float* gln = (const float*)d_in[10];
  const float* bln = (const float*)d_in[11];
  const float* W2  = (const float*)d_in[12];
  const float* b2  = (const float*)d_in[13];

  float* out = (float*)d_out;
  float* out0   = out;                       // [4,2048,256]
  float* out1   = out + 2097152;             // [4,2048,256]
  float* attn01 = out + 4194304;             // [4,4,2048,2048]
  float* attn10 = out + 4194304 + 67108864;  // [4,4,2048,2048]

  // ---- workspace layout (bf16 unless noted); 256B aligned chunks
  char* ws = (char*)d_ws;
  size_t off = 0;
  auto take = [&](size_t bytes) { size_t o = off; off += (bytes + 255) & ~(size_t)255; return o; };
  const size_t TOK = 8192;                       // tokens per stream
  bf16* wqk_b = (bf16*)(ws + take(65536 * 2));
  bf16* wv_b  = (bf16*)(ws + take(65536 * 2));
  bf16* wo_b  = (bf16*)(ws + take(65536 * 2));
  bf16* w1_b  = (bf16*)(ws + take(262144 * 2));
  bf16* w2_b  = (bf16*)(ws + take(131072 * 2));
  size_t qk0_o = take(TOK * 256 * 2);            // 4 contiguous 4MB chunks ->
  size_t qk1_o = take(TOK * 256 * 2);            //   reused as hpre0 (16MB f32)
  size_t v0_o  = take(TOK * 256 * 2);
  size_t v1_o  = take(TOK * 256 * 2);
  size_t v0T_o = take(TOK * 256 * 2);            // 4 contiguous 4MB chunks ->
  size_t v1T_o = take(TOK * 256 * 2);            //   reused as hpre1 (16MB f32)
  size_t m0_o  = take(TOK * 256 * 2);
  size_t m1_o  = take(TOK * 256 * 2);
  size_t cat0_o = take(TOK * 512 * 2);           // reused as h0 (bf16)
  size_t cat1_o = take(TOK * 512 * 2);           // reused as h1 (bf16)
  (void)qk1_o; (void)v1T_o; (void)m0_o; (void)m1_o;
  bf16* qk0b = (bf16*)(ws + qk0_o);
  bf16* qk1b = (bf16*)(ws + qk1_o);
  bf16* v0b  = (bf16*)(ws + v0_o);
  bf16* v1b  = (bf16*)(ws + v1_o);
  bf16* v0T  = (bf16*)(ws + v0T_o);
  bf16* v1T  = (bf16*)(ws + v1T_o);
  bf16* m0b  = (bf16*)(ws + m0_o);
  bf16* m1b  = (bf16*)(ws + m1_o);
  bf16* cat0 = (bf16*)(ws + cat0_o);
  bf16* cat1 = (bf16*)(ws + cat1_o);
  float* hpre0 = (float*)(ws + qk0_o);           // alias (qk/v dead by then)
  float* hpre1 = (float*)(ws + v0T_o);           // alias (vT/m dead by then)
  bf16* h0 = cat0;                               // alias (cat dead by then)
  bf16* h1 = cat1;

  const float mult = 0.35355339059327379f;       // 64^-0.25
  dim3 blk(256);

  // 1) weights -> bf16
  cvt_f32_bf16<<<256, blk, 0, stream>>>(Wqk, wqk_b, 65536);
  cvt_f32_bf16<<<256, blk, 0, stream>>>(Wv,  wv_b,  65536);
  cvt_f32_bf16<<<256, blk, 0, stream>>>(Wo,  wo_b,  65536);
  cvt_f32_bf16<<<1024, blk, 0, stream>>>(W1, w1_b, 262144);
  cvt_f32_bf16<<<512, blk, 0, stream>>>(W2,  w2_b, 131072);
  // 2) x -> cat[:, 0:256]
  cvt_rows_bf16<<<8192, blk, 0, stream>>>(x0, cat0);
  cvt_rows_bf16<<<8192, blk, 0, stream>>>(x1, cat1);

  // 3) projections: qk = (x Wqk^T + bqk)*mult ; v = x Wv^T + bv   (A is fp32)
  dim3 gp(4, 64, 1);
  gemm_wmma<true><<<gp, blk, 0, stream>>>(x0, 256, 0, wqk_b, 256, 0, bqk, mult,
                                          nullptr, qk0b, nullptr, 256, 0, 0, 1, 256);
  gemm_wmma<true><<<gp, blk, 0, stream>>>(x1, 256, 0, wqk_b, 256, 0, bqk, mult,
                                          nullptr, qk1b, nullptr, 256, 0, 0, 1, 256);
  gemm_wmma<true><<<gp, blk, 0, stream>>>(x0, 256, 0, wv_b, 256, 0, bv, 1.0f,
                                          nullptr, v0b, nullptr, 256, 0, 0, 1, 256);
  gemm_wmma<true><<<gp, blk, 0, stream>>>(x1, 256, 0, wv_b, 256, 0, bv, 1.0f,
                                          nullptr, v1b, nullptr, 256, 0, 0, 1, 256);
  // 4) per-head V transpose for attn.V GEMM B-fragments
  transpose_v<<<8192, blk, 0, stream>>>(v0b, v0T);
  transpose_v<<<8192, blk, 0, stream>>>(v1b, v1T);

  // 5) sim -> raw scores into attn01 and attn10 (transposed)
  sim_wmma<<<dim3(32, 16, 16), blk, 0, stream>>>(qk0b, qk1b, attn01, attn10);
  // 6) softmax over last axis of both buffers (65536 rows of 2048), in place
  softmax_rows<<<65536, blk, 0, stream>>>(attn01);

  // 7) m0 = attn01 . v1 ; m1 = attn10 . v0   (per-head batched, A fp32)
  dim3 gm(1, 16, 16);
  gemm_wmma<true><<<gm, blk, 0, stream>>>(attn01, 2048, 4194304LL, v1T, 2048, 131072,
                                          nullptr, 1.0f, nullptr, m0b, nullptr,
                                          256, 524288, 64, 4, 2048);
  gemm_wmma<true><<<gm, blk, 0, stream>>>(attn10, 2048, 4194304LL, v0T, 2048, 131072,
                                          nullptr, 1.0f, nullptr, m1b, nullptr,
                                          256, 524288, 64, 4, 2048);

  // 8) Wo linear -> cat[:, 256:512]
  gemm_wmma<false><<<gp, blk, 0, stream>>>(m0b, 256, 0, wo_b, 256, 0, bo, 1.0f,
                                           nullptr, cat0 + 256, nullptr, 512, 0, 0, 1, 256);
  gemm_wmma<false><<<gp, blk, 0, stream>>>(m1b, 256, 0, wo_b, 256, 0, bo, 1.0f,
                                           nullptr, cat1 + 256, nullptr, 512, 0, 0, 1, 256);

  // 9) W1 linear -> hpre (fp32)
  dim3 g1(8, 64, 1);
  gemm_wmma<false><<<g1, blk, 0, stream>>>(cat0, 512, 0, w1_b, 512, 0, b1, 1.0f,
                                           hpre0, nullptr, nullptr, 512, 0, 0, 1, 512);
  gemm_wmma<false><<<g1, blk, 0, stream>>>(cat1, 512, 0, w1_b, 512, 0, b1, 1.0f,
                                           hpre1, nullptr, nullptr, 512, 0, 0, 1, 512);
  // 10) LN + exact GELU -> h (bf16)
  ln_gelu<<<8192, blk, 0, stream>>>(hpre0, gln, bln, h0);
  ln_gelu<<<8192, blk, 0, stream>>>(hpre1, gln, bln, h1);

  // 11) W2 linear + residual -> d_out
  gemm_wmma<false><<<gp, blk, 0, stream>>>(h0, 512, 0, w2_b, 512, 0, b2, 1.0f,
                                           out0, nullptr, x0, 256, 0, 0, 1, 512);
  gemm_wmma<false><<<gp, blk, 0, stream>>>(h1, 512, 0, w2_b, 512, 0, b2, 1.0f,
                                           out1, nullptr, x1, 256, 0, 0, 1, 512);
}
